// GeneDecoder_63513976373345
// MI455X (gfx1250) — matrix operand order repaired and checked by reading, not compiled, not run on hardware
//
#include <hip/hip_runtime.h>

// Problem constants (match reference)
#define NN   20000   // nodes
#define EE   100000  // edges
#define INF_ 256     // input feature dim
#define DD   512     // per-head dim
#define HH   4       // heads
#define HD   2048    // H*D

// ---------------- WMMA types ----------------
typedef __attribute__((ext_vector_type(16))) __bf16 v16bf;
typedef __attribute__((ext_vector_type(8)))  __bf16 v8bf;
typedef __attribute__((ext_vector_type(8)))  float  v8f;

union Frag16 { v16bf v; v8bf h[2]; };

__device__ __forceinline__ v8bf load8(const __bf16* p) {
    return *reinterpret_cast<const v8bf*>(p);
}

// One pipeline stage of fragments: two A row-tiles (32 rows) + four B col-tiles (64 cols)
struct Frags { Frag16 a0, a1, b[4]; };

__device__ __forceinline__ void load_frags(Frags& f,
        const __bf16* __restrict__ Ar0, const __bf16* __restrict__ Ar1,
        const __bf16* __restrict__ Br, int ldBt, int k)
{
    f.a0.h[0] = load8(Ar0 + k);
    f.a0.h[1] = load8(Ar0 + k + 16);
    f.a1.h[0] = load8(Ar1 + k);
    f.a1.h[1] = load8(Ar1 + k + 16);
#pragma unroll
    for (int j = 0; j < 4; ++j) {
        const __bf16* Bp = Br + (size_t)j * 16 * ldBt + k;
        f.b[j].h[0] = load8(Bp);
        f.b[j].h[1] = load8(Bp + 16);
    }
}

__device__ __forceinline__ void do_wmma(const Frags& f, v8f acc0[4], v8f acc1[4])
{
#pragma unroll
    for (int j = 0; j < 4; ++j) {
        acc0[j] = __builtin_amdgcn_wmma_f32_16x16x32_bf16(
                      false, f.a0.v, false, f.b[j].v, (short)0, acc0[j], false, false);
        acc1[j] = __builtin_amdgcn_wmma_f32_16x16x32_bf16(
                      false, f.a1.v, false, f.b[j].v, (short)0, acc1[j], false, false);
    }
}

// ---------------- helpers ----------------
__device__ __forceinline__ unsigned mapf(float f) {
    unsigned u = __float_as_uint(f);
    return (u & 0x80000000u) ? ~u : (u | 0x80000000u);
}
__device__ __forceinline__ float unmapf(unsigned m) {
    unsigned u = (m & 0x80000000u) ? (m ^ 0x80000000u) : ~m;
    return __uint_as_float(u);
}

// ---------------- prep kernels ----------------
__global__ void k_f32_to_bf16(const float* __restrict__ in, __bf16* __restrict__ out, int n) {
    int i = blockIdx.x * blockDim.x + threadIdx.x;
    if (i < n) out[i] = (__bf16)in[i];
}

// W is [K][Ncol] row-major fp32; Wt is [Ncol][K] row-major bf16
__global__ void k_transpose_bf16(const float* __restrict__ W, __bf16* __restrict__ Wt, int K, int Ncol) {
    int i = blockIdx.x * blockDim.x + threadIdx.x;
    if (i >= K * Ncol) return;
    int n = i / K;
    int k = i - n * K;
    Wt[i] = (__bf16)W[(size_t)k * Ncol + n];
}

// bias2[n] = b_fc[n] + sum_k gat_bias[k] * W_fc[k][n]
__global__ void k_fuse_bias(const float* __restrict__ Wfc, const float* __restrict__ gb,
                            const float* __restrict__ bfc, float* __restrict__ bias2) {
    int n = blockIdx.x * blockDim.x + threadIdx.x;
    if (n >= DD) return;
    float s = bfc[n];
    for (int k = 0; k < HD; ++k) s += gb[k] * Wfc[(size_t)k * DD + n];
    bias2[n] = s;
}

__global__ void k_fill_u32(unsigned* __restrict__ p, unsigned v, int n) {
    int i = blockIdx.x * blockDim.x + threadIdx.x;
    if (i < n) p[i] = v;
}

// ---------------- bf16 WMMA GEMM ----------------
// C[M x Ncols] = A[M x K] (bf16, row-major, ldA) * Bt[Ncols x K]^T (bf16, ldBt) + bias[col]
// grid.x = M/32 row tiles, grid.y = Ncols/512 col super-tiles, block = 256 (8 waves).
// Each wave computes a 32x64 tile: 2 A frags x 4 B frags = 8 WMMA per 32-wide k-step,
// with double-buffered register fragments for load/WMMA overlap.
__global__ __launch_bounds__(256) void k_gemm_bf16(
        const __bf16* __restrict__ A, const __bf16* __restrict__ Bt,
        const float* __restrict__ bias, float* __restrict__ C,
        int K, int ldA, int ldBt, int ldC, int doLeaky)
{
    const int w    = threadIdx.x >> 5;
    const int lane = threadIdx.x & 31;
    const int half = lane >> 4;     // wave halves select K sub-chunks per ISA layout
    const int ln   = lane & 15;

    const int r0 = blockIdx.x * 32;
    const int c0 = blockIdx.y * 512 + w * 64;

    // A fragments: row = r0+ln (+16) ; element chunks at K = k + half*8 and +16
    const __bf16* Ar0 = A + (size_t)(r0 + ln) * ldA + half * 8;
    const __bf16* Ar1 = Ar0 + (size_t)16 * ldA;
    // B fragments: col = c0 + j*16 + ln ; element chunks at K = k + half*8 and +16
    const __bf16* Br  = Bt + (size_t)(c0 + ln) * ldBt + half * 8;

    v8f acc0[4] = {};
    v8f acc1[4] = {};

    const int nk = K >> 5;          // k-steps of 32; even for all our GEMMs
    Frags f0, f1;
    load_frags(f0, Ar0, Ar1, Br, ldBt, 0);
    for (int i = 0; i < nk; i += 2) {
        // prefetch step i+1, compute step i
        load_frags(f1, Ar0, Ar1, Br, ldBt, (i + 1) << 5);
        do_wmma(f0, acc0, acc1);
        // prefetch step i+2 (if any), compute step i+1
        if (i + 2 < nk) load_frags(f0, Ar0, Ar1, Br, ldBt, (i + 2) << 5);
        do_wmma(f1, acc0, acc1);
    }

#pragma unroll
    for (int j = 0; j < 4; ++j) {
        const int col = c0 + j * 16 + ln;
        const float bv = bias[col];
#pragma unroll
        for (int e = 0; e < 8; ++e) {
            const int row = r0 + e + half * 8;   // C/D layout: VGPR e -> M = e (+8 upper half)
            float v0 = acc0[j][e] + bv;
            float v1 = acc1[j][e] + bv;
            if (doLeaky) {
                v0 = v0 > 0.f ? v0 : 0.2f * v0;
                v1 = v1 > 0.f ? v1 : 0.2f * v1;
            }
            C[(size_t)row * ldC + col]        = v0;
            C[(size_t)(row + 16) * ldC + col] = v1;
        }
    }
}

// ---------------- edge kernels ----------------
// one block (4 waves) per edge; wave h handles head h
__global__ __launch_bounds__(128) void k_edge_scores(
        const float* __restrict__ fs, const float* __restrict__ fd,
        const int* __restrict__ src, const int* __restrict__ dst,
        const float* __restrict__ attn, float* __restrict__ score,
        unsigned* __restrict__ nodeMaxU)
{
    const int e = blockIdx.x;
    const int h = threadIdx.x >> 5;
    const int lane = threadIdx.x & 31;
    const int s = src[e], d = dst[e];
    const float* ps = fs + (size_t)s * HD + h * DD;
    const float* pd = fd + (size_t)d * HD + h * DD;
    const float* pa = attn + h * DD;
    float sum = 0.f;
    for (int i = lane; i < DD; i += 32) {
        float v = ps[i] + pd[i];
        v = v > 0.f ? v : 0.2f * v;
        sum += pa[i] * v;
    }
#pragma unroll
    for (int off = 16; off; off >>= 1) sum += __shfl_xor(sum, off, 32);
    if (lane == 0) {
        score[(size_t)e * HH + h] = sum;
        __hip_atomic_fetch_max(&nodeMaxU[(size_t)d * HH + h], mapf(sum),
                               __ATOMIC_RELAXED, __HIP_MEMORY_SCOPE_AGENT);
    }
}

__global__ void k_edge_exp(float* __restrict__ score, const int* __restrict__ dst,
                           const unsigned* __restrict__ nodeMaxU, float* __restrict__ denom)
{
    int idx = blockIdx.x * blockDim.x + threadIdx.x;
    if (idx >= EE * HH) return;
    int e = idx >> 2, h = idx & 3;
    int d = dst[e];
    float m = unmapf(nodeMaxU[(size_t)d * HH + h]);
    float ex = expf(score[idx] - m);
    score[idx] = ex;   // reuse buffer for exp values
    __hip_atomic_fetch_add(&denom[(size_t)d * HH + h], ex,
                           __ATOMIC_RELAXED, __HIP_MEMORY_SCOPE_AGENT);
}

// one block (256 threads) per edge; each thread handles 8 contiguous floats (one head)
__global__ __launch_bounds__(256) void k_aggregate(
        const float* __restrict__ fs, const int* __restrict__ src, const int* __restrict__ dst,
        const float* __restrict__ ex, const float* __restrict__ denom, float* __restrict__ rst)
{
    const int e = blockIdx.x;
    const int t = threadIdx.x;
    const int s = src[e], d = dst[e];
    const int h = t >> 6;                 // (t*8)/512
    const float alpha = ex[(size_t)e * HH + h] / denom[(size_t)d * HH + h];
    const float4* ps = reinterpret_cast<const float4*>(fs + (size_t)s * HD) + t * 2;
    float* pr = rst + (size_t)d * HD + t * 8;
    float4 x0 = ps[0], x1 = ps[1];
    float vals[8] = { x0.x, x0.y, x0.z, x0.w, x1.x, x1.y, x1.z, x1.w };
#pragma unroll
    for (int i = 0; i < 8; ++i)
        __hip_atomic_fetch_add(&pr[i], alpha * vals[i],
                               __ATOMIC_RELAXED, __HIP_MEMORY_SCOPE_AGENT);
}

// ---------------- launch ----------------
extern "C" void kernel_launch(void* const* d_in, const int* in_sizes, int n_in,
                              void* d_out, int out_size, void* d_ws, size_t ws_size,
                              hipStream_t stream) {
    (void)in_sizes; (void)n_in; (void)out_size; (void)ws_size;
    const float* z        = (const float*)d_in[0];
    const int*   src      = (const int*)  d_in[1];
    const int*   dst      = (const int*)  d_in[2];
    const float* W_src    = (const float*)d_in[3];
    const float* b_src    = (const float*)d_in[4];
    const float* W_dst    = (const float*)d_in[5];
    const float* b_dst    = (const float*)d_in[6];
    const float* attn     = (const float*)d_in[7];
    const float* gat_bias = (const float*)d_in[8];
    const float* W_fc     = (const float*)d_in[9];
    const float* b_fc     = (const float*)d_in[10];
    float* out = (float*)d_out;

    // workspace carve-out (~344 MB)
    size_t off = 0;
    char* ws = (char*)d_ws;
    auto carve = [&](size_t bytes) {
        void* p = ws + off;
        off += (bytes + 255) & ~(size_t)255;
        return p;
    };
    __bf16*  zb       = (__bf16*) carve((size_t)NN * INF_ * 2);
    __bf16*  wsrcT    = (__bf16*) carve((size_t)HD * INF_ * 2);
    __bf16*  wdstT    = (__bf16*) carve((size_t)HD * INF_ * 2);
    __bf16*  wfcT     = (__bf16*) carve((size_t)DD * HD * 2);
    float*   bias2    = (float*)  carve((size_t)DD * 4);
    unsigned* nodeMaxU= (unsigned*)carve((size_t)NN * HH * 4);
    float*   denom    = (float*)  carve((size_t)NN * HH * 4);
    float*   score    = (float*)  carve((size_t)EE * HH * 4);
    float*   feat_src = (float*)  carve((size_t)NN * HD * 4);
    float*   feat_dst = (float*)  carve((size_t)NN * HD * 4);  // reused as rst
    float*   rst      = feat_dst;
    __bf16*  rstb     = (__bf16*)feat_src;  // alias: feat_src dead after aggregate

    const int T = 256;

    // prep: bf16 conversions / transposes / fused bias
    k_f32_to_bf16<<<(NN * INF_ + T - 1) / T, T, 0, stream>>>(z, zb, NN * INF_);
    k_transpose_bf16<<<(INF_ * HD + T - 1) / T, T, 0, stream>>>(W_src, wsrcT, INF_, HD);
    k_transpose_bf16<<<(INF_ * HD + T - 1) / T, T, 0, stream>>>(W_dst, wdstT, INF_, HD);
    k_transpose_bf16<<<(HD * DD + T - 1) / T, T, 0, stream>>>(W_fc, wfcT, HD, DD);
    k_fuse_bias<<<(DD + T - 1) / T, T, 0, stream>>>(W_fc, gat_bias, b_fc, bias2);

    // feat_src = z @ W_src + b_src ; feat_dst = z @ W_dst + b_dst
    k_gemm_bf16<<<dim3(NN / 32, HD / 512), T, 0, stream>>>(
        zb, wsrcT, b_src, feat_src, INF_, INF_, INF_, HD, 0);
    k_gemm_bf16<<<dim3(NN / 32, HD / 512), T, 0, stream>>>(
        zb, wdstT, b_dst, feat_dst, INF_, INF_, INF_, HD, 0);

    // edge softmax state
    k_fill_u32<<<(NN * HH + T - 1) / T, T, 0, stream>>>(nodeMaxU, 0x007FFFFFu, NN * HH); // map(-inf)
    k_fill_u32<<<(NN * HH + T - 1) / T, T, 0, stream>>>((unsigned*)denom, 0u, NN * HH);

    // per-edge scores + running max
    k_edge_scores<<<EE, 128, 0, stream>>>(feat_src, feat_dst, src, dst, attn, score, nodeMaxU);
    // exp + denominator
    k_edge_exp<<<(EE * HH + T - 1) / T, T, 0, stream>>>(score, dst, nodeMaxU, denom);

    // rst = scatter-add alpha * feat_src[src]  (rst reuses feat_dst buffer, zero first)
    k_fill_u32<<<(NN * HD + T - 1) / T, T, 0, stream>>>((unsigned*)rst, 0u, NN * HD);
    k_aggregate<<<EE, T, 0, stream>>>(feat_src, src, dst, score, denom, rst);

    // out = leaky_relu(rst @ W_fc + (b_fc + gat_bias @ W_fc))
    k_f32_to_bf16<<<(NN * HD + T - 1) / T, T, 0, stream>>>(rst, rstb, NN * HD);
    k_gemm_bf16<<<dim3(NN / 32, 1), T, 0, stream>>>(
        rstb, wfcT, bias2, out, HD, HD, HD, DD, 1);
}
